// M3GNetGraphConv_876173328556
// MI455X (gfx1250) — compile-verified, matrix operand order checked by reading, not run on hardware
//
#include <hip/hip_runtime.h>

#define FEAT   64
#define INDIM  192
#define TILE_E 128
#define NTHR   256
#define DEG    9

typedef __attribute__((ext_vector_type(16))) __bf16 v16bf;
typedef __attribute__((ext_vector_type(8)))  __bf16 v8bf;
typedef __attribute__((ext_vector_type(8)))  float  v8f;

union Frag16 { v16bf v; v8bf h[2]; };

// dynamic LDS layout (bytes)
#define SX_OFF 0                      // [128][192] bf16 : x = [vi|vj|e]
#define SW_OFF 49152                  // [64][192]  bf16 : pre-transposed weights
#define SH_OFF 73728                  // [128][64]  bf16 : layers-branch hidden
#define SG_OFF 90112                  // [128][64]  bf16 : gates-branch hidden
#define SR_OFF 106496                 // [128][9]   f32  : rbf tile
#define LDS_BYTES 111104

// bf16 workspace layout (elements) — transposed weights [n][k]
#define WS_EW1 0
#define WS_EG1 12288
#define WS_EW2 24576
#define WS_EG2 28672
#define WS_NW1 32768
#define WS_NG1 45056
#define WS_NW2 57344
#define WS_NG2 61440

__device__ __forceinline__ __bf16 f2bf(float f) {
  union { float f; unsigned u; } x; x.f = f;
  unsigned r = x.u + 0x7fffu + ((x.u >> 16) & 1u);   // RNE truncate
  unsigned short hs = (unsigned short)(r >> 16);
  __bf16 b; __builtin_memcpy(&b, &hs, sizeof(b));
  return b;
}

__device__ __forceinline__ float softplus2f(float x) {
  return fmaxf(x, 0.0f) + __logf(1.0f + __expf(-fabsf(x))) - 0.69314718055994531f;
}
__device__ __forceinline__ float sigmoidf(float x) {
  return 1.0f / (1.0f + __expf(-x));
}

// CDNA5 async global->LDS bulk copy (ASYNCcnt path), 16B granules.
// vdst = per-lane LDS byte address, vaddr = per-lane global byte offset,
// saddr = uniform 64-bit base (GVS mode).
__device__ __forceinline__ void async_copy_to_lds(__bf16* dst, const __bf16* src,
                                                  int nbf16, int tid) {
  const unsigned lds0 = (unsigned)(uintptr_t)dst;          // low 32 bits = LDS offset
  const unsigned long long g0 = (unsigned long long)(uintptr_t)src;
  const int chunks = nbf16 >> 3;                            // 8 bf16 = 16 B
  for (int ch = tid; ch < chunks; ch += NTHR) {
    unsigned l   = lds0 + ((unsigned)ch << 4);
    unsigned off = (unsigned)ch << 4;
    asm volatile("global_load_async_to_lds_b128 %0, %1, %2"
                 :: "v"(l), "v"(off), "s"(g0) : "memory");
  }
}
__device__ __forceinline__ void wait_async() {
  asm volatile("s_wait_asynccnt 0x0" ::: "memory");
}

// Load one 16x32 bf16 fragment from LDS tile (row-major, `stride` elems).
// 16-bit A layout: lanes 0-15 -> K chunks {0..7,16..23}; lanes 16-31 -> {8..15,24..31}.
__device__ __forceinline__ v16bf ldfrag(const __bf16* base, int tile0, int stride,
                                        int kbase, int lane) {
  const int r  = tile0 + (lane & 15);
  const int c0 = kbase + ((lane >> 4) << 3);
  Frag16 f;
  f.h[0] = *(const v8bf*)(base + r * stride + c0);
  f.h[1] = *(const v8bf*)(base + r * stride + c0 + 16);
  return f.v;
}

__device__ __forceinline__ v8f wmma_bf16(v16bf a, v16bf b, v8f c) {
  return __builtin_amdgcn_wmma_f32_16x16x32_bf16(false, a, false, b, (short)0, c,
                                                 false, false);
}

// One gated-MLP phase over the 128-edge tile (weights pre-transposed bf16).
template <bool EDGE_PHASE>
__device__ __forceinline__ void gated_phase(
    __bf16* sx, __bf16* sw, __bf16* sh, __bf16* sg, const float* srbf,
    const __bf16* __restrict__ W1t, const float* __restrict__ b1,
    const __bf16* __restrict__ W2t, const float* __restrict__ b2,
    const __bf16* __restrict__ G1t, const float* __restrict__ gb1,
    const __bf16* __restrict__ G2t, const float* __restrict__ gb2,
    const float* __restrict__ Wrbf, const float* __restrict__ edge_feat,
    const int* s_dst, int e0, int E,
    float* __restrict__ e_out, float* __restrict__ v_out,
    int tid, int lane, int wave)
{
  // ---------- layer 1, layers branch: sh = bf16(softplus2(x @ W1 + b1)) ----------
  async_copy_to_lds(sw, W1t, INDIM * FEAT, tid);
  wait_async();
  __syncthreads();
  for (int t = wave; t < 32; t += 8) {
    const int mt = (t >> 2) << 4, nt = (t & 3) << 4;
    v8f c = {0.f, 0.f, 0.f, 0.f, 0.f, 0.f, 0.f, 0.f};
    #pragma unroll
    for (int kb = 0; kb < INDIM; kb += 32) {
      v16bf a = ldfrag(sx, mt, INDIM, kb, lane);
      v16bf b = ldfrag(sw, nt, INDIM, kb, lane);
      c = wmma_bf16(a, b, c);
    }
    const int row = mt + ((lane >> 4) << 3);
    const int col = nt + (lane & 15);
    const float bias = b1[col];
    #pragma unroll
    for (int r = 0; r < 8; ++r)
      sh[(row + r) * FEAT + col] = f2bf(softplus2f(c[r] + bias));
  }
  __syncthreads();

  // ---------- layer 1, gates branch: sg = bf16(softplus2(x @ G1 + gb1)) ----------
  async_copy_to_lds(sw, G1t, INDIM * FEAT, tid);
  wait_async();
  __syncthreads();
  for (int t = wave; t < 32; t += 8) {
    const int mt = (t >> 2) << 4, nt = (t & 3) << 4;
    v8f c = {0.f, 0.f, 0.f, 0.f, 0.f, 0.f, 0.f, 0.f};
    #pragma unroll
    for (int kb = 0; kb < INDIM; kb += 32) {
      v16bf a = ldfrag(sx, mt, INDIM, kb, lane);
      v16bf b = ldfrag(sw, nt, INDIM, kb, lane);
      c = wmma_bf16(a, b, c);
    }
    const int row = mt + ((lane >> 4) << 3);
    const int col = nt + (lane & 15);
    const float bias = gb1[col];
    #pragma unroll
    for (int r = 0; r < 8; ++r)
      sg[(row + r) * FEAT + col] = f2bf(softplus2f(c[r] + bias));
  }
  __syncthreads();

  // ---------- layer 2, layers branch: h2 in registers ----------
  async_copy_to_lds(sw, W2t, FEAT * FEAT, tid);
  wait_async();
  __syncthreads();
  float h2[4][8];
  #pragma unroll
  for (int tt = 0; tt < 4; ++tt) {
    const int t = wave + (tt << 3);
    const int mt = (t >> 2) << 4, nt = (t & 3) << 4;
    v8f c = {0.f, 0.f, 0.f, 0.f, 0.f, 0.f, 0.f, 0.f};
    #pragma unroll
    for (int kb = 0; kb < FEAT; kb += 32) {
      v16bf a = ldfrag(sh, mt, FEAT, kb, lane);
      v16bf b = ldfrag(sw, nt, FEAT, kb, lane);
      c = wmma_bf16(a, b, c);
    }
    const float bias = b2[nt + (lane & 15)];
    #pragma unroll
    for (int r = 0; r < 8; ++r) h2[tt][r] = softplus2f(c[r] + bias);
  }
  __syncthreads();

  // ---------- layer 2, gates branch + fused epilogue ----------
  async_copy_to_lds(sw, G2t, FEAT * FEAT, tid);
  wait_async();
  __syncthreads();
  #pragma unroll
  for (int tt = 0; tt < 4; ++tt) {
    const int t = wave + (tt << 3);
    const int mt = (t >> 2) << 4, nt = (t & 3) << 4;
    v8f c = {0.f, 0.f, 0.f, 0.f, 0.f, 0.f, 0.f, 0.f};
    #pragma unroll
    for (int kb = 0; kb < FEAT; kb += 32) {
      v16bf a = ldfrag(sg, mt, FEAT, kb, lane);
      v16bf b = ldfrag(sw, nt, FEAT, kb, lane);
      c = wmma_bf16(a, b, c);
    }
    const int rowb = mt + ((lane >> 4) << 3);
    const int col  = nt + (lane & 15);
    const float bias = gb2[col];
    float wr[DEG];
    #pragma unroll
    for (int d = 0; d < DEG; ++d) wr[d] = Wrbf[d * FEAT + col];
    #pragma unroll
    for (int r = 0; r < 8; ++r) {
      const int erow = rowb + r;
      const long e = (long)e0 + erow;
      if (e < E) {
        const float g2 = sigmoidf(c[r] + bias);
        float rd = 0.f;
        #pragma unroll
        for (int d = 0; d < DEG; ++d) rd = fmaf(srbf[erow * DEG + d], wr[d], rd);
        const float m = h2[tt][r] * g2 * rd;
        if (EDGE_PHASE) {
          const float en = edge_feat[e * FEAT + col] + m;
          e_out[e * FEAT + col] = en;
          sx[erow * INDIM + 2 * FEAT + col] = f2bf(en);  // refresh x for node phase
        } else {
          atomicAdd(&v_out[(long)s_dst[erow] * FEAT + col], m);
        }
      }
    }
  }
  __syncthreads();
}

__global__ __launch_bounds__(NTHR) void m3gnet_fused(
    const float* __restrict__ edge_feat, const float* __restrict__ node_feat,
    const float* __restrict__ rbf, const int* __restrict__ src,
    const int* __restrict__ dst,
    const __bf16* __restrict__ ws,         // pre-transposed bf16 weights
    const float* __restrict__ eb1, const float* __restrict__ eb2,
    const float* __restrict__ egb1, const float* __restrict__ egb2,
    const float* __restrict__ eWrbf,
    const float* __restrict__ nb1, const float* __restrict__ nb2,
    const float* __restrict__ ngb1, const float* __restrict__ ngb2,
    const float* __restrict__ nWrbf,
    float* __restrict__ e_out, float* __restrict__ v_out, int E)
{
  extern __shared__ char smem[];
  __bf16* sx   = (__bf16*)(smem + SX_OFF);
  __bf16* sw   = (__bf16*)(smem + SW_OFF);
  __bf16* sh   = (__bf16*)(smem + SH_OFF);
  __bf16* sg   = (__bf16*)(smem + SG_OFF);
  float*  srbf = (float*)(smem + SR_OFF);
  __shared__ int s_src[TILE_E], s_dst[TILE_E];

  const int tid  = threadIdx.x;
  const int lane = tid & 31;
  const int wave = tid >> 5;
  const int e0   = blockIdx.x * TILE_E;

  for (int i = tid; i < TILE_E; i += NTHR) {
    int e = e0 + i; if (e >= E) e = E - 1;
    s_src[i] = src[e];
    s_dst[i] = dst[e];
  }
  // stage rbf tile (f32)
  for (int i = tid; i < TILE_E * DEG; i += NTHR) {
    long e = (long)e0 + i / DEG; if (e >= E) e = E - 1;
    srbf[i] = rbf[e * DEG + (i % DEG)];
  }
  // CDNA5 prefetch of the next tile's edge features (global_prefetch_b8)
  if (e0 + TILE_E < E)
    __builtin_prefetch(&edge_feat[(long)(e0 + TILE_E) * FEAT + tid], 0, 1);
  __syncthreads();

  // stage x = [vi | vj | eij] as bf16
  for (int i = tid; i < TILE_E * INDIM; i += NTHR) {
    const int r = i / INDIM, c = i % INDIM;
    long e = (long)e0 + r; if (e >= E) e = E - 1;
    float v;
    if (c < FEAT)            v = node_feat[(long)s_src[r] * FEAT + c];
    else if (c < 2 * FEAT)   v = node_feat[(long)s_dst[r] * FEAT + (c - FEAT)];
    else                     v = edge_feat[e * FEAT + (c - 2 * FEAT)];
    sx[r * INDIM + c] = f2bf(v);
  }
  __syncthreads();

  // Edge update (Eq. 4): e_new = e + gated_mlp(x)*(rbf@eWrbf); also rewrites x.
  gated_phase<true>(sx, sw, sh, sg, srbf,
                    ws + WS_EW1, eb1, ws + WS_EW2, eb2,
                    ws + WS_EG1, egb1, ws + WS_EG2, egb2,
                    eWrbf, edge_feat, s_dst, e0, E, e_out, v_out,
                    tid, lane, wave);

  // Node update (Eq. 5): scatter gated_mlp(x')*(rbf@nWrbf) into v_out[dst].
  gated_phase<false>(sx, sw, sh, sg, srbf,
                     ws + WS_NW1, nb1, ws + WS_NW2, nb2,
                     ws + WS_NG1, ngb1, ws + WS_NG2, ngb2,
                     nWrbf, edge_feat, s_dst, e0, E, e_out, v_out,
                     tid, lane, wave);
}

// Transpose + convert one [K x 64] f32 weight matrix into bf16 [64 x K].
__global__ void prep_w(const float* __restrict__ in, __bf16* __restrict__ out,
                       int K) {
  int i = blockIdx.x * NTHR + threadIdx.x;
  if (i < K * FEAT) {
    int k = i >> 6, n = i & 63;
    out[n * K + k] = f2bf(in[i]);
  }
}

__global__ void init_vout(const float* __restrict__ nf, float* __restrict__ vout,
                          int n) {
  int i = blockIdx.x * NTHR + threadIdx.x;
  if (i < n) vout[i] = nf[i];
}

extern "C" void kernel_launch(void* const* d_in, const int* in_sizes, int n_in,
                              void* d_out, int out_size, void* d_ws, size_t ws_size,
                              hipStream_t stream) {
  const float* edge_feat = (const float*)d_in[0];
  const float* node_feat = (const float*)d_in[1];
  const float* rbf       = (const float*)d_in[2];
  const int*   src       = (const int*)d_in[3];
  const int*   dst       = (const int*)d_in[4];
  const float* eW1  = (const float*)d_in[5];
  const float* eb1  = (const float*)d_in[6];
  const float* eW2  = (const float*)d_in[7];
  const float* eb2  = (const float*)d_in[8];
  const float* eG1  = (const float*)d_in[9];
  const float* egb1 = (const float*)d_in[10];
  const float* eG2  = (const float*)d_in[11];
  const float* egb2 = (const float*)d_in[12];
  const float* eWrbf= (const float*)d_in[13];
  const float* nW1  = (const float*)d_in[14];
  const float* nb1  = (const float*)d_in[15];
  const float* nW2  = (const float*)d_in[16];
  const float* nb2  = (const float*)d_in[17];
  const float* nG1  = (const float*)d_in[18];
  const float* ngb1 = (const float*)d_in[19];
  const float* nG2  = (const float*)d_in[20];
  const float* ngb2 = (const float*)d_in[21];
  const float* nWrbf= (const float*)d_in[22];

  const int E  = in_sizes[0] / FEAT;
  const int NF = in_sizes[1];
  float* e_out = (float*)d_out;
  float* v_out = (float*)d_out + (size_t)E * FEAT;
  __bf16* ws = (__bf16*)d_ws;

  // One-time weight transpose+convert into bf16 workspace (L2-resident).
  const int g1 = (INDIM * FEAT + NTHR - 1) / NTHR;
  const int g2 = (FEAT * FEAT + NTHR - 1) / NTHR;
  prep_w<<<g1, NTHR, 0, stream>>>(eW1, ws + WS_EW1, INDIM);
  prep_w<<<g1, NTHR, 0, stream>>>(eG1, ws + WS_EG1, INDIM);
  prep_w<<<g2, NTHR, 0, stream>>>(eW2, ws + WS_EW2, FEAT);
  prep_w<<<g2, NTHR, 0, stream>>>(eG2, ws + WS_EG2, FEAT);
  prep_w<<<g1, NTHR, 0, stream>>>(nW1, ws + WS_NW1, INDIM);
  prep_w<<<g1, NTHR, 0, stream>>>(nG1, ws + WS_NG1, INDIM);
  prep_w<<<g2, NTHR, 0, stream>>>(nW2, ws + WS_NW2, FEAT);
  prep_w<<<g2, NTHR, 0, stream>>>(nG2, ws + WS_NG2, FEAT);

  // v_new starts as node_feat; main kernel atomically accumulates messages.
  init_vout<<<(NF + NTHR - 1) / NTHR, NTHR, 0, stream>>>(node_feat, v_out, NF);

  const int tiles = (E + TILE_E - 1) / TILE_E;
  m3gnet_fused<<<tiles, NTHR, LDS_BYTES, stream>>>(
      edge_feat, node_feat, rbf, src, dst, ws,
      eb1, eb2, egb1, egb2, eWrbf,
      nb1, nb2, ngb1, ngb2, nWrbf,
      e_out, v_out, E);
}
